// ProphetEncoder_70669391888755
// MI455X (gfx1250) — compile-verified, matrix operand order checked by reading, not run on hardware
//
#include <hip/hip_runtime.h>
#include <math.h>

// ---------------------------------------------------------------------------
// CDNA5 (gfx1250) wave32 WMMA transformer encoder forward.
// Matmuls: v_wmma_f32_16x16x32_f16. GEMM weight panels staged into LDS with
// the Tensor Data Mover (tensor_load_to_lds, TENSORcnt) -- with a
// global_load_async_to_lds_b128 (ASYNCcnt) path as runtime-selectable backup.
// ---------------------------------------------------------------------------

typedef _Float16 v16h __attribute__((ext_vector_type(16)));
typedef _Float16 v8h  __attribute__((ext_vector_type(8)));
typedef float    v8f  __attribute__((ext_vector_type(8)));
typedef unsigned int v4u __attribute__((ext_vector_type(4)));
typedef int          v8i __attribute__((ext_vector_type(8)));
typedef int          v4i __attribute__((ext_vector_type(4)));

#define DEVFN static __device__ __forceinline__

DEVFN v8f wmma_f16(v16h a, v16h b, v8f c) {
  // D = A(16x32 f16) x B(32x16 f16) + C(16x16 f32)
  return __builtin_amdgcn_wmma_f32_16x16x32_f16(false, a, false, b, (short)0, c,
                                                false, false);
}

// Load one 16-bit A/B fragment row for this lane. `rowk` points at the start
// of this lane's (row, k-base) element. Per ISA 16-bit layout:
//   lanes 0-15 : K = kb+0..7  and kb+16..23
//   lanes 16-31: K = kb+8..15 and kb+24..31
template <typename P>
DEVFN v16h load_frag(P rowk, int lane) {
  const int off = (lane >> 4) << 3;
  v8h lo = *(const v8h*)(rowk + off);
  v8h hi = *(const v8h*)(rowk + off + 16);
  v16h r;
#pragma unroll
  for (int i = 0; i < 8; ++i) { r[i] = lo[i]; r[i + 8] = hi[i]; }
  return r;
}

DEVFN float gelu_f(float x) {  // tanh approximation (jax.nn.gelu default)
  return 0.5f * x * (1.0f + tanhf(0.7978845608028654f * (x + 0.044715f * x * x * x)));
}

// Async 16-byte global -> LDS copy (gfx1250, tracked by ASYNCcnt).
DEVFN void async_g2l_b128(unsigned lds_off, const void* gptr) {
  asm volatile("global_load_async_to_lds_b128 %0, %1, off"
               :: "v"(lds_off), "v"((unsigned long long)gptr)
               : "memory");
}
DEVFN void wait_asynccnt0() {
  asm volatile("s_wait_asynccnt 0" ::: "memory");
}

// TDM: copy nbytes (multiple of 8, <= 64KB) from global to LDS as a single
// tensor_load_to_lds (D# per CDNA5 ISA 8.3/8.4: 2D tile, data_size = 8B,
// tile_dim0 = tensor_dim0 = nbytes/8, tile_dim1 = 1, groups 2/3 unused).
// 6-arg builtin form: (v4u g0, v8i g1, v4i g2, v4i g3, v8i extra, i32 cpol).
DEVFN void tdm_load_1d(unsigned lds_byte_addr, const void* gptr, unsigned nbytes) {
  const unsigned ne = nbytes >> 3;                    // 8-byte elements
  const unsigned long long ga = (unsigned long long)gptr;
  v4u g0;
  g0[0] = 1u;                                        // count=1, user mode
  g0[1] = lds_byte_addr;                             // lds_addr
  g0[2] = (unsigned)(ga & 0xFFFFFFFFu);              // global_addr[31:0]
  g0[3] = (unsigned)((ga >> 32) & 0x1FFFFFFu)        // global_addr[56:32]
        | (2u << 30);                                // type = 2 ("image")
  v8i g1;
  g1[0] = (int)(3u << 16);                           // wg_mask=0, data_size=3 (8B)
  g1[1] = (int)((ne & 0xFFFFu) << 16);               // tensor_dim0[15:0]
  g1[2] = (int)((ne >> 16) & 0xFFFFu)                // tensor_dim0[31:16]
        | (int)(1u << 16);                           // tensor_dim1 = 1 (low 16)
  g1[3] = (int)((ne & 0xFFFFu) << 16);               // tile_dim0 = ne (<= 8192)
  g1[4] = 1;                                         // tile_dim1 = 1, tile_dim2 = 0
  g1[5] = (int)ne;                                   // tensor_dim0_stride[31:0]
  g1[6] = (int)((ne >> 16) & 0xFFFFu);               // stride[47:32]; dim1_stride=0
  g1[7] = 0;
  v4i z4 = {0, 0, 0, 0};
  v8i z8 = {0, 0, 0, 0, 0, 0, 0, 0};
  __builtin_amdgcn_tensor_load_to_lds(g0, g1, z4, z4, z8, 0);
}

// ---------------------------------------------------------------------------
// Cooperative WMMA GEMM: out = epilogue(A[MxK] @ Wt^T + bias [+ resid])
// A: f16 row-major (lda). Wt: f16 N-major (N rows of K) so A/B fragments share
// one addressing pattern. Block = 128 threads (4 waves) covering 64 rows x 64
// cols; the 64xK weight panel (<= 64KB) is staged once into LDS (TDM or async
// loads) and consumed by all 4 waves.
// mode: 0 = f32 row-major, 1 = f16 row-major,
//       3 = f16 scatter to [B,H,T,32] (Q/K), 4 = f16 scatter to [B,H,32,T] (V^T)
// act:  0 = none, 1 = gelu
// ---------------------------------------------------------------------------
__global__ __launch_bounds__(128)
void gemm16_k(const _Float16* __restrict__ A, int lda,
              const _Float16* __restrict__ Wt, const float* __restrict__ bias,
              const float* __restrict__ resid,
              float* __restrict__ outF, _Float16* __restrict__ outH,
              int M, int N, int K, int act, int mode, int use_tdm) {
  __shared__ _Float16 wlds[64 * 512];  // 64 output cols x K (K <= 512): 64 KB max
  const int tid  = threadIdx.x;
  const int lane = tid & 31;
  const int wv   = tid >> 5;                 // wave 0..3
  const int mt = blockIdx.x * 64 + wv * 16;  // this wave's 16-row tile
  const int nt = blockIdx.y * 64;

  // ---- stage weight panel Wt[nt .. nt+63][0..K) into LDS (contiguous) -----
  {
    const _Float16* src = Wt + (size_t)nt * K;
    if (use_tdm) {
      if (tid < 32) {  // wave 0 issues one TDM op for the whole panel
        tdm_load_1d((unsigned)(size_t)&wlds[0], (const void*)src,
                    (unsigned)(64 * K * 2));
      }
      __builtin_amdgcn_s_wait_tensorcnt(0);
    } else {
      const int chunks = (64 * K) >> 3;      // 16-byte chunks; multiple of 128
      for (int i = tid; i < chunks; i += 128) {
        async_g2l_b128((unsigned)(size_t)&wlds[i * 8],
                       (const void*)(src + (size_t)i * 8));
      }
      wait_asynccnt0();
    }
  }
  __syncthreads();

  int arow = mt + (lane & 15);
  if (arow >= M) arow = M - 1;  // clamp loads; stores are guarded
  const _Float16* Arow = A + (size_t)arow * lda;

  v8f acc[4];
#pragma unroll
  for (int s = 0; s < 4; ++s) acc[s] = (v8f){0.f, 0.f, 0.f, 0.f, 0.f, 0.f, 0.f, 0.f};

  for (int kb = 0; kb < K; kb += 32) {
    v16h a = load_frag(Arow + kb, lane);
#pragma unroll
    for (int s = 0; s < 4; ++s) {
      v16h b = load_frag(&wlds[(s * 16 + (lane & 15)) * K + kb], lane);
      acc[s] = wmma_f16(a, b, acc[s]);
    }
  }

#pragma unroll
  for (int s = 0; s < 4; ++s) {
#pragma unroll
    for (int r = 0; r < 8; ++r) {
      const int m = mt + r + ((lane >> 4) << 3);
      const int n = nt + s * 16 + (lane & 15);
      if (m < M) {
        float v = acc[s][r] + bias[n];
        if (resid) v += resid[(size_t)m * N + n];
        if (act == 1) v = gelu_f(v);
        if (mode == 0) {
          outF[(size_t)m * N + n] = v;
        } else if (mode == 1) {
          outH[(size_t)m * N + n] = (_Float16)v;
        } else if (mode == 3) {               // [B,H,T,32]
          const int bi = m >> 10, t = m & 1023;
          outH[((size_t)(bi * 8 + (n >> 5)) * 1024 + t) * 32 + (n & 31)] = (_Float16)v;
        } else {                              // mode 4: [B,H,32,T]
          const int bi = m >> 10, t = m & 1023;
          outH[((size_t)(bi * 8 + (n >> 5)) * 32 + (n & 31)) * 1024 + t] = (_Float16)v;
        }
      }
    }
  }
}

// ---------------------------------------------------------------------------
// Attention: one wave per (b, head, 16-query tile). T=1024 keys.
// Scores live in 64KB LDS (16 x 1024 f32); softmax normalization is folded
// into the output epilogue so probs stay f32 in LDS.
// ---------------------------------------------------------------------------
__global__ __launch_bounds__(32)
void attn_k(const _Float16* __restrict__ q16, const _Float16* __restrict__ k16,
            const _Float16* __restrict__ vt16, const unsigned char* __restrict__ mask,
            _Float16* __restrict__ ao16) {
  __shared__ float sc[16 * 1024];  // exactly 64 KB
  const int lane = threadIdx.x;
  const int qt = blockIdx.x;   // query tile 0..63
  const int hh = blockIdx.y;   // head 0..7
  const int b  = blockIdx.z;   // batch 0..3
  const _Float16* qbase = q16 + ((size_t)(b * 8 + hh) * 1024 + qt * 16) * 32;
  const _Float16* kbase = k16 + (size_t)(b * 8 + hh) * 1024 * 32;
  const _Float16* vbase = vt16 + (size_t)(b * 8 + hh) * 32 * 1024;

  // ---- S = (Q K^T) * 1/sqrt(D) -------------------------------------------
  v16h aq = load_frag(qbase + (size_t)(lane & 15) * 32, lane);
#pragma unroll 4
  for (int jt = 0; jt < 64; ++jt) {
    v16h bk = load_frag(kbase + (size_t)(jt * 16 + (lane & 15)) * 32, lane);
    v8f c = {};
    c = wmma_f16(aq, bk, c);
    const int col = jt * 16 + (lane & 15);
#pragma unroll
    for (int r = 0; r < 8; ++r)
      sc[(r + ((lane >> 4) << 3)) * 1024 + col] = c[r] * 0.0625f;  // 1/sqrt(256)
  }
  __syncthreads();

  // ---- masked softmax over each of the 16 rows ---------------------------
  const unsigned char* mrow = mask + (size_t)b * 1024;
  float inv_loc[8] = {};
#pragma unroll
  for (int i = 0; i < 16; ++i) {
    float vals[32];
    float mx = -3.0e38f;
#pragma unroll
    for (int u = 0; u < 32; ++u) {
      const int j = lane * 32 + u;
      float s = sc[i * 1024 + j];
      if (!mrow[j]) s = -3.0e38f;
      vals[u] = s;
      mx = fmaxf(mx, s);
    }
#pragma unroll
    for (int o2 = 16; o2 > 0; o2 >>= 1) mx = fmaxf(mx, __shfl_xor(mx, o2, 32));
    float sum = 0.0f;
#pragma unroll
    for (int u = 0; u < 32; ++u) {
      const float e = (vals[u] <= -3.0e38f) ? 0.0f : __expf(vals[u] - mx);
      sc[i * 1024 + lane * 32 + u] = e;
      sum += e;
    }
#pragma unroll
    for (int o2 = 16; o2 > 0; o2 >>= 1) sum += __shfl_xor(sum, o2, 32);
    const float inv = 1.0f / sum;
    if ((i < 8) == (lane < 16)) inv_loc[i & 7] = inv;  // lane's half-wave owns row i
  }
  __syncthreads();

  // ---- O = P V  (P converted to f16 fragments on the fly) ----------------
  v8f o0 = {}, o1 = {};
  const int off = (lane >> 4) << 3;
  for (int kb = 0; kb < 1024; kb += 32) {
    const float* prow = sc + (size_t)(lane & 15) * 1024 + kb + off;
    v16h ap;
#pragma unroll
    for (int i = 0; i < 8; ++i) {
      ap[i]     = (_Float16)prow[i];
      ap[i + 8] = (_Float16)prow[16 + i];
    }
    v16h b0 = load_frag(vbase + (size_t)(lane & 15) * 1024 + kb, lane);
    v16h b1 = load_frag(vbase + (size_t)(16 + (lane & 15)) * 1024 + kb, lane);
    o0 = wmma_f16(ap, b0, o0);
    o1 = wmma_f16(ap, b1, o1);
  }
  _Float16* dst = ao16 + (size_t)(b * 1024 + qt * 16) * 256 + hh * 32;
#pragma unroll
  for (int r = 0; r < 8; ++r) {
    const int row = r + ((lane >> 4) << 3);
    const float is = inv_loc[r];
    dst[(size_t)row * 256 + (lane & 15)]      = (_Float16)(o0[r] * is);
    dst[(size_t)row * 256 + 16 + (lane & 15)] = (_Float16)(o1[r] * is);
  }
}

// ---------------------------------------------------------------------------
// LayerNorm over 256-wide rows; one wave per row; writes f32 + f16 copies.
// ---------------------------------------------------------------------------
__global__ __launch_bounds__(32)
void ln_k(const float* __restrict__ in, const float* __restrict__ scale,
          const float* __restrict__ shift,
          float* __restrict__ xout, _Float16* __restrict__ x16out) {
  const int row = blockIdx.x;
  const int lane = threadIdx.x;
  const float* r = in + (size_t)row * 256;
  float v[8];
  float sum = 0.0f;
#pragma unroll
  for (int u = 0; u < 8; ++u) { v[u] = r[lane + u * 32]; sum += v[u]; }
#pragma unroll
  for (int o2 = 16; o2 > 0; o2 >>= 1) sum += __shfl_xor(sum, o2, 32);
  const float mean = sum * (1.0f / 256.0f);
  float var = 0.0f;
#pragma unroll
  for (int u = 0; u < 8; ++u) { const float d = v[u] - mean; var += d * d; }
#pragma unroll
  for (int o2 = 16; o2 > 0; o2 >>= 1) var += __shfl_xor(var, o2, 32);
  const float rstd = rsqrtf(var * (1.0f / 256.0f) + 1e-6f);
#pragma unroll
  for (int u = 0; u < 8; ++u) {
    const int c = lane + u * 32;
    const float y = (v[u] - mean) * rstd * scale[c] + shift[c];
    xout[(size_t)row * 256 + c] = y;
    x16out[(size_t)row * 256 + c] = (_Float16)y;
  }
}

// ---------------------------------------------------------------------------
// Small helper kernels.
// ---------------------------------------------------------------------------
// W (K x N, f32) -> Wt (N x K, f16)
__global__ void wcvt_k(const float* __restrict__ W, _Float16* __restrict__ Wt,
                       int K, int N) {
  const int idx = blockIdx.x * 256 + threadIdx.x;
  if (idx >= K * N) return;
  const int n = idx / K, k = idx - n * K;
  Wt[idx] = (_Float16)W[(size_t)k * N + n];
}

__global__ void cvt16_k(const float* __restrict__ s, _Float16* __restrict__ d, int n) {
  const int idx = blockIdx.x * 256 + threadIdx.x;
  if (idx < n) d[idx] = (_Float16)s[idx];
}

// Assemble x = [cls, emb[drug], tok] into f32 + f16 buffers (B*T rows of 256).
__global__ void build_x_k(const float* __restrict__ tokf, const float* __restrict__ emb,
                          const int* __restrict__ didx, const float* __restrict__ cls,
                          float* __restrict__ x, _Float16* __restrict__ x16) {
  const int idx = blockIdx.x * 256 + threadIdx.x;
  if (idx >= 4 * 1024 * 256) return;
  const int c = idx & 255;
  const int row = idx >> 8;
  const int t = row & 1023, b = row >> 10;
  float v;
  if (t == 0)      v = cls[c];
  else if (t == 1) v = emb[(size_t)didx[b] * 256 + c];
  else             v = tokf[(size_t)(b * 1022 + (t - 2)) * 256 + c];
  x[idx] = v;
  x16[idx] = (_Float16)v;
}

// out = (x[:,0,:], zeros) flattened: 2048 floats.
__global__ void final_out_k(const float* __restrict__ x, float* __restrict__ out) {
  const int idx = blockIdx.x * 256 + threadIdx.x;  // 1024 threads
  const int b = idx >> 8, c = idx & 255;
  out[idx] = x[(size_t)(b * 1024) * 256 + c];
  out[1024 + idx] = 0.0f;
}

// ---------------------------------------------------------------------------
extern "C" void kernel_launch(void* const* d_in, const int* in_sizes, int n_in,
                              void* d_out, int out_size, void* d_ws, size_t ws_size,
                              hipStream_t stream) {
  (void)in_sizes; (void)n_in; (void)out_size; (void)ws_size;

  const float* x_expr  = (const float*)d_in[0];
  const int*   drugidx = (const int*)d_in[1];
  const unsigned char* mask = (const unsigned char*)d_in[2];
  const float* tok_W1 = (const float*)d_in[3];
  const float* tok_b1 = (const float*)d_in[4];
  const float* tok_W2 = (const float*)d_in[5];
  const float* tok_b2 = (const float*)d_in[6];
  const float* embt   = (const float*)d_in[7];
  const float* cls    = (const float*)d_in[8];
  const float* Wq = (const float*)d_in[9];
  const float* bq = (const float*)d_in[10];
  const float* Wk = (const float*)d_in[11];
  const float* bk = (const float*)d_in[12];
  const float* Wv = (const float*)d_in[13];
  const float* bv = (const float*)d_in[14];
  const float* Wo = (const float*)d_in[15];
  const float* bo = (const float*)d_in[16];
  const float* ln1_s = (const float*)d_in[17];
  const float* ln1_b = (const float*)d_in[18];
  const float* Wf1 = (const float*)d_in[19];
  const float* bf1 = (const float*)d_in[20];
  const float* Wf2 = (const float*)d_in[21];
  const float* bf2 = (const float*)d_in[22];
  const float* ln2_s = (const float*)d_in[23];
  const float* ln2_b = (const float*)d_in[24];

  // ---- workspace layout ---------------------------------------------------
  char* ws = (char*)d_ws;
  size_t off = 0;
  auto take = [&](size_t bytes) -> char* {
    char* p = ws + off;
    off += (bytes + 255) & ~(size_t)255;
    return p;
  };
  float*    xb     = (float*)take((size_t)4096 * 256 * 4);
  _Float16* x16    = (_Float16*)take((size_t)4096 * 256 * 2);
  float*    residb = (float*)take((size_t)4096 * 256 * 4);
  _Float16* xe16   = (_Float16*)take((size_t)4096 * 64 * 2);
  _Float16* h16    = (_Float16*)take((size_t)4096 * 256 * 2);
  float*    tokf   = (float*)take((size_t)4096 * 256 * 4);
  _Float16* q16    = (_Float16*)take((size_t)4 * 8 * 1024 * 32 * 2);
  _Float16* k16    = (_Float16*)take((size_t)4 * 8 * 1024 * 32 * 2);
  _Float16* vt16   = (_Float16*)take((size_t)4 * 8 * 32 * 1024 * 2);
  _Float16* ao16   = (_Float16*)take((size_t)4096 * 256 * 2);
  _Float16* ffn16  = (_Float16*)take((size_t)4096 * 512 * 2);
  _Float16* wt1    = (_Float16*)take((size_t)256 * 64 * 2);
  _Float16* wt2    = (_Float16*)take((size_t)256 * 256 * 2);
  _Float16* wqt    = (_Float16*)take((size_t)4 * 256 * 256 * 2);
  _Float16* wkt    = (_Float16*)take((size_t)4 * 256 * 256 * 2);
  _Float16* wvt    = (_Float16*)take((size_t)4 * 256 * 256 * 2);
  _Float16* wot    = (_Float16*)take((size_t)4 * 256 * 256 * 2);
  _Float16* wf1t   = (_Float16*)take((size_t)4 * 512 * 256 * 2);
  _Float16* wf2t   = (_Float16*)take((size_t)4 * 256 * 512 * 2);

  const dim3 blk128(128);
  const dim3 wave(32);
  const int USE_TDM = 1;

  // ---- weight conversion (f32 KxN -> f16 NxK) ----------------------------
  wcvt_k<<<(256 * 64 + 255) / 256, 256, 0, stream>>>(tok_W1, wt1, 64, 256);
  wcvt_k<<<(256 * 256 + 255) / 256, 256, 0, stream>>>(tok_W2, wt2, 256, 256);
  for (int l = 0; l < 4; ++l) {
    wcvt_k<<<256, 256, 0, stream>>>(Wq + (size_t)l * 65536, wqt + (size_t)l * 65536, 256, 256);
    wcvt_k<<<256, 256, 0, stream>>>(Wk + (size_t)l * 65536, wkt + (size_t)l * 65536, 256, 256);
    wcvt_k<<<256, 256, 0, stream>>>(Wv + (size_t)l * 65536, wvt + (size_t)l * 65536, 256, 256);
    wcvt_k<<<256, 256, 0, stream>>>(Wo + (size_t)l * 65536, wot + (size_t)l * 65536, 256, 256);
    wcvt_k<<<512, 256, 0, stream>>>(Wf1 + (size_t)l * 131072, wf1t + (size_t)l * 131072, 256, 512);
    wcvt_k<<<512, 256, 0, stream>>>(Wf2 + (size_t)l * 131072, wf2t + (size_t)l * 131072, 512, 256);
  }

  // ---- tokenizer MLP ------------------------------------------------------
  const int ME = 4 * 1022;  // 4088 expr-token rows
  cvt16_k<<<(ME * 64 + 255) / 256, 256, 0, stream>>>(x_expr, xe16, ME * 64);
  // h = gelu(x_expr @ W1 + b1) -> f16
  gemm16_k<<<dim3(64, 4), blk128, 0, stream>>>(xe16, 64, wt1, tok_b1, nullptr,
                                               nullptr, h16, ME, 256, 64, 1, 1, USE_TDM);
  // tok = gelu(h @ W2 + b2) -> f32
  gemm16_k<<<dim3(64, 4), blk128, 0, stream>>>(h16, 256, wt2, tok_b2, nullptr,
                                               tokf, nullptr, ME, 256, 256, 1, 0, USE_TDM);
  // x = concat(cls, emb[drug], tok)
  build_x_k<<<4096, 256, 0, stream>>>(tokf, embt, drugidx, cls, xb, x16);

  // ---- transformer layers -------------------------------------------------
  for (int l = 0; l < 4; ++l) {
    const size_t wo = (size_t)l * 65536;
    // Q, K -> [B,H,T,32]; V -> [B,H,32,T]
    gemm16_k<<<dim3(64, 4), blk128, 0, stream>>>(x16, 256, wqt + wo, bq + l * 256,
                                                 nullptr, nullptr, q16, 4096, 256, 256, 0, 3, USE_TDM);
    gemm16_k<<<dim3(64, 4), blk128, 0, stream>>>(x16, 256, wkt + wo, bk + l * 256,
                                                 nullptr, nullptr, k16, 4096, 256, 256, 0, 3, USE_TDM);
    gemm16_k<<<dim3(64, 4), blk128, 0, stream>>>(x16, 256, wvt + wo, bv + l * 256,
                                                 nullptr, nullptr, vt16, 4096, 256, 256, 0, 4, USE_TDM);
    attn_k<<<dim3(64, 8, 4), wave, 0, stream>>>(q16, k16, vt16, mask, ao16);
    // resid = x + ao @ Wo + bo
    gemm16_k<<<dim3(64, 4), blk128, 0, stream>>>(ao16, 256, wot + wo, bo + l * 256,
                                                 xb, residb, nullptr, 4096, 256, 256, 0, 0, USE_TDM);
    ln_k<<<4096, wave, 0, stream>>>(residb, ln1_s + l * 256, ln1_b + l * 256, xb, x16);
    // ffn = gelu(x @ Wf1 + bf1) -> f16
    gemm16_k<<<dim3(64, 8), blk128, 0, stream>>>(x16, 256, wf1t + (size_t)l * 131072,
                                                 bf1 + l * 512, nullptr, nullptr, ffn16,
                                                 4096, 512, 256, 1, 1, USE_TDM);
    // resid = x + ffn @ Wf2 + bf2
    gemm16_k<<<dim3(64, 4), blk128, 0, stream>>>(ffn16, 512, wf2t + (size_t)l * 131072,
                                                 bf2 + l * 256, xb, residb, nullptr,
                                                 4096, 256, 512, 0, 0, USE_TDM);
    ln_k<<<4096, wave, 0, stream>>>(residb, ln2_s + l * 256, ln2_b + l * 256, xb, x16);
  }

  final_out_k<<<4, 256, 0, stream>>>(xb, (float*)d_out);
}
